// LU_M2SR_64372969832972
// MI455X (gfx1250) — matrix operand order repaired and checked by reading, not compiled, not run on hardware
//
#include <hip/hip_runtime.h>
#include <hip/hip_bf16.h>
#include <math.h>

typedef __attribute__((ext_vector_type(16))) _Float16 v16h;
typedef __attribute__((ext_vector_type(8)))  float    v8f;

#define BATCH   16
#define CFEAT   64
#define LSEQ    16384            // 128*128
#define GTOK    8
#define DINNER  16
#define DSTATE  16
#define NHEADS_ 2
#define HEADDIM 8
#define DCONV   3
#define CONVDIM 48               // DINNER + 2*DSTATE
#define DPROJ   66               // 2*DINNER + 2*DSTATE + NHEADS
#define LC      16               // chunk length (one WMMA tile)
#define NCH     (LSEQ / LC)      // 1024 chunks per batch

__device__ __forceinline__ float silu_f(float x)     { return x / (1.f + __expf(-x)); }
__device__ __forceinline__ float softplus_f(float x) { return (x > 20.f) ? x : log1pf(__expf(x)); }

// ---- WMMA fragment loaders, wave32: contiguous k order (k = 16h + e), same for A and B ----
// A (16x32 f16, row-major [m][k]): lane m=lane&15, kb = (lane&16)?16:0,
//   elems e -> k=kb+e
__device__ __forceinline__ v16h lds_load_a16(const _Float16* T, int lane) {
  const int m  = lane & 15;
  const int kb = (lane & 16) ? 16 : 0;
  v16h a;
#pragma unroll
  for (int e = 0; e < 8; ++e) {
    a[e]     = T[m * 32 + kb + e];
    a[8 + e] = T[m * 32 + kb + 8 + e];
  }
  return a;
}
// B (32x16 f16, row-major [k][n]): lane n=lane&15, kb=(lane&16)?16:0, elems e -> k=kb+e
__device__ __forceinline__ v16h lds_load_b16(const _Float16* T, int lane) {
  const int n  = lane & 15;
  const int kb = (lane & 16) ? 16 : 0;
  v16h b;
#pragma unroll
  for (int e = 0; e < 16; ++e) b[e] = T[(kb + e) * 16 + n];
  return b;
}

// ============ Kernel 1: tokenize (64->8) + in_proj (8->66), channel-major outputs ============
__global__ __launch_bounds__(256) void k_tok_inproj(
    const float* __restrict__ x, const float* __restrict__ tok_w, const float* __restrict__ tok_b,
    const float* __restrict__ ipw, const float* __restrict__ ipb,
    float* __restrict__ gt, float* __restrict__ zx)
{
  __shared__ float sw[GTOK * CFEAT];
  __shared__ float siw[DPROJ * GTOK];
  __shared__ float sib[DPROJ];
  __shared__ float sb[GTOK];
  for (int i = threadIdx.x; i < GTOK * CFEAT; i += 256) sw[i] = tok_w[i];
  for (int i = threadIdx.x; i < DPROJ * GTOK; i += 256) siw[i] = ipw[i];
  if (threadIdx.x < DPROJ) sib[threadIdx.x] = ipb[threadIdx.x];
  if (threadIdx.x < GTOK)  sb[threadIdx.x]  = tok_b[threadIdx.x];
  __syncthreads();

  const int gid = blockIdx.x * 256 + threadIdx.x;          // over BATCH*LSEQ
  const int b = gid / LSEQ, l = gid % LSEQ;
  const float* xb = x + (size_t)b * CFEAT * LSEQ + l;

  float acc[GTOK];
#pragma unroll
  for (int g = 0; g < GTOK; ++g) acc[g] = sb[g];
  for (int c = 0; c < CFEAT; ++c) {
    const float xv = xb[(size_t)c * LSEQ];
#pragma unroll
    for (int g = 0; g < GTOK; ++g) acc[g] += xv * sw[g * CFEAT + c];
  }
#pragma unroll
  for (int g = 0; g < GTOK; ++g)
    gt[((size_t)b * GTOK + g) * LSEQ + l] = acc[g];

  for (int k = 0; k < DPROJ; ++k) {
    float v = sib[k];
#pragma unroll
    for (int g = 0; g < GTOK; ++g) v += siw[k * GTOK + g] * acc[g];
    zx[((size_t)b * DPROJ + k) * LSEQ + l] = v;
  }
}

// ============ Kernel 2: per-chunk intra scan via WMMA (1 wave per 16-token chunk) ============
__global__ __launch_bounds__(32) void k_chunk(
    const float* __restrict__ zx, const float* __restrict__ conv_w, const float* __restrict__ conv_b,
    const float* __restrict__ A_log, const float* __restrict__ Dp, const float* __restrict__ dt_bias,
    float* __restrict__ Yg, float* __restrict__ Ptg, float* __restrict__ Hc, float* __restrict__ Pp)
{
  __shared__ float    xconv[LC][CONVDIM];   // conv'd + silu'd xBC
  __shared__ float    dts[LC][NHEADS_];     // softplus dt
  __shared__ float    Pcum[NHEADS_][LC];    // cumulative decay products
  __shared__ float    Wd[NHEADS_][LC];      // dts[s]/Pcum[s]  (division-free masks)
  __shared__ _Float16 Ct[16][32];           // A-tile: C (t x n), K padded
  __shared__ _Float16 Bt[32][16];           // B-tile: B^T (n x s), K padded
  __shared__ _Float16 St[16][32];           // A-tile: masked scores (t x s)
  __shared__ _Float16 Xt[32][16];           // B-tile: X_h (s x p)
  __shared__ _Float16 Xw[16][32];           // A-tile: weighted X_h^T (p x s)
  __shared__ _Float16 Bw[32][16];           // B-tile: B (s x n)

  const int lane  = threadIdx.x;
  const int chunk = blockIdx.x % NCH;
  const int b     = blockIdx.x / NCH;
  const int l0    = chunk * LC;
  const float* zxb = zx + (size_t)b * DPROJ * LSEQ;

  // inline causal depthwise conv (k=3) + silu over 48 channels x 16 positions
  for (int idx = lane; idx < LC * CONVDIM; idx += 32) {
    const int t = idx / CONVDIM, ch = idx % CONVDIM;
    const float* row = zxb + (size_t)(DINNER + ch) * LSEQ;
    const int l = l0 + t;
    float acc = conv_b[ch];
#pragma unroll
    for (int j = 0; j < DCONV; ++j) {
      const int ls = l - (DCONV - 1) + j;
      acc += ((ls >= 0) ? row[ls] : 0.f) * conv_w[ch * DCONV + j];
    }
    xconv[t][ch] = silu_f(acc);
  }
  for (int idx = lane; idx < LC * NHEADS_; idx += 32) {
    const int t = idx / NHEADS_, h = idx % NHEADS_;
    dts[t][h] = softplus_f(zxb[(size_t)(DINNER + CONVDIM + h) * LSEQ + l0 + t] + dt_bias[h]);
  }
  __syncthreads();

  if (lane < NHEADS_) {
    const float Ae = __expf(A_log[lane]);
    float p = 1.f;
    for (int t = 0; t < LC; ++t) {
      p *= __expf(-Ae * dts[t][lane]);
      Pcum[lane][t] = p;
      Ptg[((size_t)b * NHEADS_ + lane) * LSEQ + l0 + t] = p;
    }
    Pp[((size_t)b * NCH + chunk) * NHEADS_ + lane] = p;
  }
  __syncthreads();

  // one division per lane for the whole chunk: W[h][s] = dt_s / P_s
  {
    const int h = lane >> 4, s = lane & 15;
    Wd[h][s] = dts[s][h] / Pcum[h][s];
  }

  // build head-independent tiles: Ct (A), Bt = B^T (B for C@B^T), Bw = B (B for state GEMM)
  for (int idx = lane; idx < 16 * 32; idx += 32) {
    const int r = idx >> 5, k = idx & 31;
    Ct[r][k] = (k < DSTATE) ? (_Float16)xconv[r][32 + k] : (_Float16)0.f;
  }
  for (int idx = lane; idx < 32 * 16; idx += 32) {
    const int r = idx >> 4, c = idx & 15;
    Bt[r][c] = (r < DSTATE) ? (_Float16)xconv[c][16 + r] : (_Float16)0.f;
    Bw[r][c] = (r < LC)     ? (_Float16)xconv[r][16 + c] : (_Float16)0.f;
  }
  __syncthreads();

  const v16h aC = lds_load_a16(&Ct[0][0], lane);
  const v16h bB = lds_load_b16(&Bt[0][0], lane);
  v8f zf = {};
  // CB[t][s] = sum_n C_t[n] * B_s[n]
  const v8f CB = __builtin_amdgcn_wmma_f32_16x16x32_f16(false, aC, false, bB, (short)0, zf, false, false);

  const int dn = lane & 15;               // D-frag N index
  const int mb = (lane & 16) ? 8 : 0;     // D-frag M base

  for (int h = 0; h < NHEADS_; ++h) {
    const float PL = Pcum[h][LC - 1];
    const float Ws = Wd[h][dn];           // per-lane column weight
    // masked score tile S[t][s] = CB * P_t * (dt_s/P_s), causal, branchless
#pragma unroll
    for (int i = 0; i < 8; ++i) {
      const int t = mb + i, s = dn;
      const float msk = (t >= s) ? 1.f : 0.f;
      const float v = CB[i] * msk * Pcum[h][t] * Ws;
      St[t][s]      = (_Float16)v;
      St[t][s + 16] = (_Float16)0.f;
    }
    for (int idx = lane; idx < 32 * 16; idx += 32) {
      const int r = idx >> 4, p = idx & 15;
      Xt[r][p] = (r < LC && p < HEADDIM) ? (_Float16)xconv[r][h * HEADDIM + p] : (_Float16)0.f;
    }
    for (int idx = lane; idx < 16 * 32; idx += 32) {
      const int p = idx >> 5, s = idx & 31;
      const float v = (p < HEADDIM && s < LC)
                        ? PL * Wd[h][s] * xconv[s][h * HEADDIM + p] : 0.f;
      Xw[p][s] = (_Float16)v;
    }
    __syncthreads();

    const v16h aS = lds_load_a16(&St[0][0], lane);
    const v16h bX = lds_load_b16(&Xt[0][0], lane);
    const v8f  Yd = __builtin_amdgcn_wmma_f32_16x16x32_f16(false, aS, false, bX, (short)0, zf, false, false);
    const v16h aW = lds_load_a16(&Xw[0][0], lane);
    const v16h bW = lds_load_b16(&Bw[0][0], lane);
    const v8f  Hd = __builtin_amdgcn_wmma_f32_16x16x32_f16(false, aW, false, bW, (short)0, zf, false, false);

    const float Dh = Dp[h];
#pragma unroll
    for (int i = 0; i < 8; ++i) {
      const int t = mb + i, p = dn;
      if (p < HEADDIM)
        Yg[((size_t)b * LSEQ + l0 + t) * DINNER + h * HEADDIM + p] = Yd[i] + Dh * xconv[t][h * HEADDIM + p];
    }
#pragma unroll
    for (int i = 0; i < 8; ++i) {
      const int p = mb + i, n = dn;
      if (p < HEADDIM)
        Hc[(((size_t)(b * NCH + chunk) * NHEADS_ + h) * HEADDIM + p) * DSTATE + n] = Hd[i];
    }
    __syncthreads();
  }
}

// ============ Kernel 3: cross-chunk prefix scan of 128-float states ============
__global__ __launch_bounds__(128) void k_scan(
    const float* __restrict__ Hc, const float* __restrict__ Pp, float* __restrict__ Hin)
{
  const int b = blockIdx.x / NHEADS_, h = blockIdx.x % NHEADS_;
  const int t = threadIdx.x;                  // p*16+n element of the state
  float hst = 0.f;
  for (int c = 0; c < NCH; ++c) {
    const size_t base = ((size_t)(b * NCH + c) * NHEADS_ + h) * (HEADDIM * DSTATE);
    Hin[base + t] = hst;                      // state entering chunk c
    const float p = Pp[((size_t)b * NCH + c) * NHEADS_ + h];
    hst = p * hst + Hc[base + t];
  }
}

// ============ Kernel 4: add state term, gate, RMSNorm, out-proj, +gt residual ============
__global__ __launch_bounds__(128) void k_finish(
    const float* __restrict__ zx, const float* __restrict__ gt,
    const float* __restrict__ Yg, const float* __restrict__ Ptg, const float* __restrict__ Hin,
    const float* __restrict__ conv_w, const float* __restrict__ conv_b,
    const float* __restrict__ norm_w, const float* __restrict__ out_w, const float* __restrict__ out_b,
    float* __restrict__ g2)
{
  const int gid = blockIdx.x * 128 + threadIdx.x;       // over BATCH*LSEQ
  if (gid >= BATCH * LSEQ) return;
  const int b = gid / LSEQ, l = gid % LSEQ;
  const int chunk = l / LC;
  const float* zxb = zx + (size_t)b * DPROJ * LSEQ;

  // recompute conv'd C channels (channels 32..47 of xBC)
  float Cv[DSTATE];
#pragma unroll
  for (int n = 0; n < DSTATE; ++n) {
    const int ch = 32 + n;
    const float* row = zxb + (size_t)(DINNER + ch) * LSEQ;
    float acc = conv_b[ch];
#pragma unroll
    for (int j = 0; j < DCONV; ++j) {
      const int ls = l - (DCONV - 1) + j;
      acc += ((ls >= 0) ? row[ls] : 0.f) * conv_w[ch * DCONV + j];
    }
    Cv[n] = silu_f(acc);
  }

  float y[DINNER];
#pragma unroll
  for (int h = 0; h < NHEADS_; ++h) {
    const float P = Ptg[((size_t)b * NHEADS_ + h) * LSEQ + l];
    const float* hi = Hin + (((size_t)(b * NCH + chunk) * NHEADS_ + h) * HEADDIM * DSTATE);
#pragma unroll
    for (int p = 0; p < HEADDIM; ++p) {
      float s = 0.f;
#pragma unroll
      for (int n = 0; n < DSTATE; ++n) s += Cv[n] * hi[p * DSTATE + n];
      y[h * HEADDIM + p] = Yg[((size_t)b * LSEQ + l) * DINNER + h * HEADDIM + p] + P * s;
    }
  }
  float ss = 0.f;
#pragma unroll
  for (int i = 0; i < DINNER; ++i) {
    const float z = zxb[(size_t)i * LSEQ + l];
    y[i] *= silu_f(z);
    ss += y[i] * y[i];
  }
  const float r = rsqrtf(ss * (1.f / DINNER) + 1e-5f);
#pragma unroll
  for (int i = 0; i < DINNER; ++i) y[i] *= r * norm_w[i];

#pragma unroll
  for (int g = 0; g < GTOK; ++g) {
    float o = out_b[g];
#pragma unroll
    for (int i = 0; i < DINNER; ++i) o += out_w[g * DINNER + i] * y[i];
    g2[((size_t)b * GTOK + g) * LSEQ + l] = o + gt[((size_t)b * GTOK + g) * LSEQ + l];
  }
}

// ============ Kernel 5: detokenize (8->64) + residual into output ============
__global__ __launch_bounds__(256) void k_detok(
    const float* __restrict__ x, const float* __restrict__ g2,
    const float* __restrict__ w, const float* __restrict__ bias, float* __restrict__ out)
{
  const size_t gid = (size_t)blockIdx.x * 256 + threadIdx.x;  // over BATCH*CFEAT*LSEQ
  const int l = (int)(gid % LSEQ);
  const int c = (int)((gid / LSEQ) % CFEAT);
  const int b = (int)(gid / ((size_t)LSEQ * CFEAT));
  float acc = x[gid] + bias[c];
#pragma unroll
  for (int g = 0; g < GTOK; ++g)
    acc += w[c * GTOK + g] * g2[((size_t)b * GTOK + g) * LSEQ + l];
  out[gid] = acc;
}

extern "C" void kernel_launch(void* const* d_in, const int* in_sizes, int n_in,
                              void* d_out, int out_size, void* d_ws, size_t ws_size,
                              hipStream_t stream) {
  (void)in_sizes; (void)n_in; (void)out_size; (void)ws_size;
  const float* x       = (const float*)d_in[0];
  const float* tok_w   = (const float*)d_in[1];
  const float* tok_b   = (const float*)d_in[2];
  const float* detok_w = (const float*)d_in[3];
  const float* detok_b = (const float*)d_in[4];
  const float* ipw     = (const float*)d_in[5];
  const float* ipb     = (const float*)d_in[6];
  const float* conv_w  = (const float*)d_in[7];
  const float* conv_b  = (const float*)d_in[8];
  const float* A_log   = (const float*)d_in[9];
  const float* Dp      = (const float*)d_in[10];
  const float* dt_bias = (const float*)d_in[11];
  const float* norm_w  = (const float*)d_in[12];
  const float* out_w   = (const float*)d_in[13];
  const float* out_b   = (const float*)d_in[14];
  float* out = (float*)d_out;

  float* ws = (float*)d_ws;
  size_t o = 0;
  float* ws_gt  = ws + o; o += (size_t)BATCH * GTOK * LSEQ;                       // 2.10M
  float* ws_zx  = ws + o; o += (size_t)BATCH * DPROJ * LSEQ;                      // 17.3M
  float* ws_Y   = ws + o; o += (size_t)BATCH * LSEQ * DINNER;                     // 4.19M
  float* ws_Pt  = ws + o; o += (size_t)BATCH * NHEADS_ * LSEQ;                    // 0.52M
  float* ws_Hc  = ws + o; o += (size_t)BATCH * NCH * NHEADS_ * HEADDIM * DSTATE;  // 4.19M
  float* ws_Hin = ws + o; o += (size_t)BATCH * NCH * NHEADS_ * HEADDIM * DSTATE;  // 4.19M
  float* ws_Pp  = ws + o; o += (size_t)BATCH * NCH * NHEADS_;                     // 33K
  float* ws_g2  = ws + o; o += (size_t)BATCH * GTOK * LSEQ;                       // 2.10M

  k_tok_inproj<<<(BATCH * LSEQ) / 256, 256, 0, stream>>>(x, tok_w, tok_b, ipw, ipb, ws_gt, ws_zx);
  k_chunk<<<BATCH * NCH, 32, 0, stream>>>(ws_zx, conv_w, conv_b, A_log, Dp, dt_bias,
                                          ws_Y, ws_Pt, ws_Hc, ws_Pp);
  k_scan<<<BATCH * NHEADS_, 128, 0, stream>>>(ws_Hc, ws_Pp, ws_Hin);
  k_finish<<<(BATCH * LSEQ) / 128, 128, 0, stream>>>(ws_zx, ws_gt, ws_Y, ws_Pt, ws_Hin,
                                                     conv_w, conv_b, norm_w, out_w, out_b, ws_g2);
  k_detok<<<(BATCH * CFEAT * LSEQ) / 256, 256, 0, stream>>>(x, ws_g2, detok_w, detok_b, out);
}
